// RCNN_36129264894534
// MI455X (gfx1250) — compile-verified, matrix operand order checked by reading, not run on hardware
//
#include <hip/hip_runtime.h>
#include <hip/hip_bf16.h>

// ---------------------------------------------------------------------------
// Problem constants
// ---------------------------------------------------------------------------
enum { Bb = 64, Ss = 512, Dd = 256, Hh = 512, Cc = 10 };

typedef __attribute__((ext_vector_type(16))) __bf16         v16bf;
typedef __attribute__((ext_vector_type(8)))  float          v8f;
typedef __attribute__((ext_vector_type(4)))  float          v4f;
typedef __attribute__((ext_vector_type(8)))  unsigned short us8;
typedef __attribute__((ext_vector_type(4)))  unsigned int   u32x4;
typedef __attribute__((ext_vector_type(8)))  int            i32x8;
typedef __attribute__((ext_vector_type(4)))  int            i32x4;

__device__ __forceinline__ v8f vzero8() {
  v8f z;
#pragma unroll
  for (int i = 0; i < 8; ++i) z[i] = 0.0f;
  return z;
}

// float -> bf16 round-to-nearest-even (bit trick)
__device__ __forceinline__ unsigned short f2bf(float f) {
  unsigned int x = __float_as_uint(f);
  x += 0x7fffu + ((x >> 16) & 1u);
  return (unsigned short)(x >> 16);
}

// Load a 16x32 bf16 tile (A-operand layout per CDNA5 ISA 7.12.2) from a
// row-major bf16 (ushort) matrix. Works for global or LDS source (address
// space inferred after inlining). Per lane: two 16-byte loads.
__device__ __forceinline__ v16bf load_tile16(const unsigned short* __restrict__ p,
                                             int ld, int lane) {
  const int m = lane & 15;
  const int h = lane >> 4;          // K-half select
  const unsigned short* q = p + (size_t)m * ld + h * 8;
  union U { us8 h2[2]; v16bf v; } u;
  u.h2[0] = *(const us8*)(q);
  u.h2[1] = *(const us8*)(q + 16);
  return u.v;
}

__device__ __forceinline__ v8f wmma_bf16(v16bf a, v16bf b, v8f c) {
  // D = A x B + C, 16x16x32, bf16 in / f32 accumulate
  return __builtin_amdgcn_wmma_f32_16x16x32_bf16(false, a, false, b,
                                                 (short)0, c, false, false);
}

// ---------------------------------------------------------------------------
// Kernel 1: convert weights to bf16, zero the pooled buffer
// ---------------------------------------------------------------------------
__global__ __launch_bounds__(256) void k_prep(
    const float* __restrict__ Wl, const float* __restrict__ Wsl,
    const float* __restrict__ Wr, const float* __restrict__ Wsr,
    const float* __restrict__ conv_w,
    unsigned short* __restrict__ Wl_bf, unsigned short* __restrict__ Wsl_bf,
    unsigned short* __restrict__ Wr_bf, unsigned short* __restrict__ Wsr_bf,
    unsigned short* __restrict__ convw_bf, float* __restrict__ pooled) {
  const int i = blockIdx.x * 256 + threadIdx.x;
  if (i < Dd * Dd) {
    Wl_bf[i]  = f2bf(Wl[i]);
    Wsl_bf[i] = f2bf(Wsl[i]);
    Wr_bf[i]  = f2bf(Wr[i]);
    Wsr_bf[i] = f2bf(Wsr[i]);
  }
  if (i < Hh * 3 * Dd) convw_bf[i] = f2bf(conv_w[i]);
  if (i < Bb * Hh) pooled[i] = 0.0f;
}

// ---------------------------------------------------------------------------
// Kernel 2: embedding gather + bf16 convert   e_bf[B,S,D]
// ---------------------------------------------------------------------------
__global__ __launch_bounds__(256) void k_embed(const int* __restrict__ x,
                                               const float* __restrict__ embed_w,
                                               unsigned short* __restrict__ e_bf) {
  const int i = blockIdx.x * 256 + threadIdx.x;   // over B*S*D = 8.4M
  const int row = i >> 8;                         // (b*S + s)
  const int d   = i & 255;
  const int tok = x[row];
  e_bf[i] = f2bf(embed_w[(size_t)tok * Dd + d]);
}

// ---------------------------------------------------------------------------
// Kernel 3: pre_{l,r}[s,d,b] = e[b,s,:] @ Ws{l,r}^T + bs{l,r} + b{l,r}
// Output layout [S, D, B] (batch contiguous) so the scan can read it with
// b128 loads.  Both biases are folded here so the scan epilogue is add-free.
// A-panel (64x256 bf16) is staged into LDS by the Tensor Data Mover with
// hardware padding producing a bank-conflict-free 264-ushort pitch.
// grid (512, 2): x = 64-row M-block, y = {left,right}
// ---------------------------------------------------------------------------
__global__ __launch_bounds__(256) void k_pre(
    const unsigned short* __restrict__ e_bf,
    const unsigned short* __restrict__ Wsl_bf,
    const unsigned short* __restrict__ Wsr_bf,
    const float* __restrict__ bsl, const float* __restrict__ bsr,
    const float* __restrict__ bl,  const float* __restrict__ br,
    float* __restrict__ pre_l, float* __restrict__ pre_r) {
  __shared__ __attribute__((aligned(16))) unsigned short atile[64 * 264];
  const int lane = threadIdx.x & 31;
  const int wave = threadIdx.x >> 5;
  const unsigned short* W = blockIdx.y ? Wsr_bf : Wsl_bf;
  const float* bias0      = blockIdx.y ? bsr : bsl;
  const float* bias1      = blockIdx.y ? br  : bl;
  float* out              = blockIdx.y ? pre_r : pre_l;
  const int m0 = blockIdx.x * 64;
  const int n0 = wave * 32;

  // ---- TDM: 2D tile load e_bf[m0:m0+64, 0:256] -> LDS (padded pitch) ----
  if (wave == 0) {
    const unsigned long long ga =
        (unsigned long long)(const void*)(e_bf + (size_t)m0 * Dd);
    const unsigned int lds = (unsigned int)(unsigned long long)(const void*)atile;
    u32x4 g0;
    g0[0] = 1u;                                            // count=1
    g0[1] = lds;                                           // lds_addr
    g0[2] = (unsigned int)(ga & 0xffffffffu);              // global_addr lo
    g0[3] = (unsigned int)((ga >> 32) & 0x1ffffffu) | (2u << 30);  // hi | type=2
    i32x8 g1;
    // data_size=2B, pad_enable, pad_interval=128 DW (512B), pad_amount=4 DW (16B)
    g1[0] = (int)((1u << 16) | (1u << 20) | (6u << 22) | (3u << 25));
    g1[1] = (int)(256u << 16);       // tensor_dim0 = 256 (lo16 in [63:48])
    g1[2] = (int)(32768u << 16);     // tensor_dim0 hi=0 | tensor_dim1 lo16 = 32768
    g1[3] = (int)(256u << 16);       // tensor_dim1 hi=0 | tile_dim0 = 256
    g1[4] = 64;                      // tile_dim1 = 64, tile_dim2 = 0
    g1[5] = 256;                     // tensor_dim0_stride = 256 (lo32)
    g1[6] = 0;                       // stride hi | tensor_dim1_stride lo
    g1[7] = 0;
    i32x4 gz;  gz[0] = 0; gz[1] = 0; gz[2] = 0; gz[3] = 0;
    i32x8 gz8;
#pragma unroll
    for (int q = 0; q < 8; ++q) gz8[q] = 0;
    __builtin_amdgcn_tensor_load_to_lds(g0, g1, gz, gz, gz8, 0);
    __builtin_amdgcn_s_wait_tensorcnt(0);
  }
  __syncthreads();

  v8f acc[4][2];
#pragma unroll
  for (int mt = 0; mt < 4; ++mt) { acc[mt][0] = vzero8(); acc[mt][1] = vzero8(); }

#pragma unroll
  for (int k = 0; k < Dd; k += 32) {
    v16bf a[4];
#pragma unroll
    for (int mt = 0; mt < 4; ++mt)
      a[mt] = load_tile16(atile + (mt * 16) * 264 + k, 264, lane);
#pragma unroll
    for (int nt = 0; nt < 2; ++nt) {
      v16bf bt = load_tile16(W + (size_t)(n0 + nt * 16) * Dd + k, Dd, lane);
#pragma unroll
      for (int mt = 0; mt < 4; ++mt) acc[mt][nt] = wmma_bf16(a[mt], bt, acc[mt][nt]);
    }
  }

  const int half = lane >> 4, nl = lane & 15;
  const int bb  = m0 >> 9;        // uniform: 64 rows share one batch element
  const int ss0 = m0 & 511;
#pragma unroll
  for (int nt = 0; nt < 2; ++nt) {
    const int col = n0 + nt * 16 + nl;
    const float bv = bias0[col] + bias1[col];
#pragma unroll
    for (int mt = 0; mt < 4; ++mt) {
#pragma unroll
      for (int r = 0; r < 8; ++r) {
        const int ss = ss0 + mt * 16 + half * 8 + r;
        out[((size_t)ss * Dd + col) * Bb + bb] = acc[mt][nt][r] + bv;
      }
    }
  }
}

// ---------------------------------------------------------------------------
// Kernel 4: sequential recurrences. grid.x = 2 (0:left fwd, 1:right bwd).
// W slice (32 cols x 256 K) register-resident per wave (16 B-tiles = 128 VGPR)
// so the 511-step dependence chain touches only LDS + WMMA + pre (b128 reads).
// carry [64 x 256] bf16 in LDS, 264-ushort pitch (conflict-free ds_b128).
// ---------------------------------------------------------------------------
__global__ __launch_bounds__(256) void k_scan(
    const unsigned short* __restrict__ Wl_bf, const unsigned short* __restrict__ Wr_bf,
    const float* __restrict__ pre_l, const float* __restrict__ pre_r,
    const float* __restrict__ cl0, const float* __restrict__ cr0,
    unsigned short* __restrict__ left_bf, unsigned short* __restrict__ right_bf) {
  __shared__ __attribute__((aligned(16))) unsigned short carry[64 * 264];

  const int dir = blockIdx.x;
  const unsigned short* W = dir ? Wr_bf : Wl_bf;
  const float* pre        = dir ? pre_r : pre_l;   // [S, D, B], biases folded
  const float* c0         = dir ? cr0 : cl0;
  unsigned short* out     = dir ? right_bf : left_bf;

  const int lane = threadIdx.x & 31;
  const int wave = threadIdx.x >> 5;
  const int n0   = wave * 32;
  const int half = lane >> 4, nl = lane & 15;

  // Hoist this wave's W columns into registers (lives across the whole scan).
  v16bf Wt[8][2];
#pragma unroll
  for (int kk = 0; kk < 8; ++kk)
#pragma unroll
    for (int nt = 0; nt < 2; ++nt)
      Wt[kk][nt] = load_tile16(W + (size_t)(n0 + nt * 16) * Dd + kk * 32, Dd, lane);

  // init: carry = c0, out[:, t_init, :] = bf16(c0)
  const int t_init = dir ? (Ss - 1) : 0;
  for (int i = threadIdx.x; i < Bb * Dd; i += 256) {
    const int bb = i >> 8, d = i & 255;
    const unsigned short v = f2bf(c0[i]);
    carry[bb * 264 + d] = v;
    out[((size_t)bb * Ss + t_init) * Dd + d] = v;
  }
  __syncthreads();

  for (int step = 0; step < Ss - 1; ++step) {
    const int ct = dir ? (Ss - 1 - step) : step;   // time the carry holds
    const int tn = dir ? (ct - 1) : (ct + 1);      // time being produced

    v8f acc[4][2];
#pragma unroll
    for (int mt = 0; mt < 4; ++mt) { acc[mt][0] = vzero8(); acc[mt][1] = vzero8(); }

#pragma unroll
    for (int kk = 0; kk < 8; ++kk) {
#pragma unroll
      for (int mt = 0; mt < 4; ++mt) {
        v16bf a = load_tile16(carry + (mt * 16) * 264 + kk * 32, 264, lane);  // LDS
        acc[mt][0] = wmma_bf16(a, Wt[kk][0], acc[mt][0]);
        acc[mt][1] = wmma_bf16(a, Wt[kk][1], acc[mt][1]);
      }
    }
    __syncthreads();   // everyone done reading old carry

#pragma unroll
    for (int nt = 0; nt < 2; ++nt) {
      const int col = n0 + nt * 16 + nl;
#pragma unroll
      for (int mt = 0; mt < 4; ++mt) {
        const int r0 = mt * 16 + half * 8;                       // first batch row
        const float* pp = pre + ((size_t)ct * Dd + col) * Bb + r0;
        const v4f p0 = *(const v4f*)pp;
        const v4f p1 = *(const v4f*)(pp + 4);
#pragma unroll
        for (int r = 0; r < 8; ++r) {
          const int bbq = r0 + r;
          float v = acc[mt][nt][r] + (r < 4 ? p0[r & 3] : p1[r & 3]);
          v = v > 0.0f ? v : 0.0f;
          const unsigned short u = f2bf(v);
          carry[bbq * 264 + col] = u;
          out[((size_t)bbq * Ss + tn) * Dd + col] = u;
        }
      }
    }
    __syncthreads();   // new carry visible before next step reads
  }
}

// ---------------------------------------------------------------------------
// Kernel 5: conv GEMM (ctx[B*S,768] @ conv_w^T) fused with max-pool over S.
// grid (512, 2): x = 64-row M-block (single batch element), y = N half.
// ---------------------------------------------------------------------------
__global__ __launch_bounds__(256) void k_conv(
    const unsigned short* __restrict__ left_bf,
    const unsigned short* __restrict__ e_bf,
    const unsigned short* __restrict__ right_bf,
    const unsigned short* __restrict__ convw_bf,
    const float* __restrict__ conv_b, float* __restrict__ pooled) {
  const int lane = threadIdx.x & 31;
  const int wave = threadIdx.x >> 5;
  const int m0 = blockIdx.x * 64;              // 64 rows, all same batch b
  const int bb = m0 >> 9;
  const int n0 = blockIdx.y * 256 + wave * 32;

  v8f acc[4][2];
#pragma unroll
  for (int mt = 0; mt < 4; ++mt) { acc[mt][0] = vzero8(); acc[mt][1] = vzero8(); }

#pragma unroll
  for (int kk = 0; kk < 24; ++kk) {
    const int k = kk * 32;
    const unsigned short* src = (kk < 8) ? left_bf : ((kk < 16) ? e_bf : right_bf);
    const int kloc = k - ((kk < 8) ? 0 : ((kk < 16) ? 256 : 512));
    v16bf a[4];
#pragma unroll
    for (int mt = 0; mt < 4; ++mt)
      a[mt] = load_tile16(src + (size_t)(m0 + mt * 16) * Dd + kloc, Dd, lane);
#pragma unroll
    for (int nt = 0; nt < 2; ++nt) {
      v16bf bt = load_tile16(convw_bf + (size_t)(n0 + nt * 16) * (3 * Dd) + k,
                             3 * Dd, lane);
#pragma unroll
      for (int mt = 0; mt < 4; ++mt) acc[mt][nt] = wmma_bf16(a[mt], bt, acc[mt][nt]);
    }
  }

  // max over the 64 rows this block owns, then atomic-max into pooled[b, col]
#pragma unroll
  for (int nt = 0; nt < 2; ++nt) {
    v8f m = acc[0][nt];
#pragma unroll
    for (int mt = 1; mt < 4; ++mt)
#pragma unroll
      for (int r = 0; r < 8; ++r) m[r] = fmaxf(m[r], acc[mt][nt][r]);
    float rmax = m[0];
#pragma unroll
    for (int r = 1; r < 8; ++r) rmax = fmaxf(rmax, m[r]);
    rmax = fmaxf(rmax, __shfl_xor(rmax, 16, 32));  // merge lane-halves (same col)
    if (lane < 16) {
      const int col = n0 + nt * 16 + lane;
      float v = rmax + conv_b[col];
      v = v > 0.0f ? v : 0.0f;                     // relu; pooled pre-zeroed
      atomicMax((int*)(pooled + (size_t)bb * Hh + col), __float_as_int(v));
    }
  }
}

// ---------------------------------------------------------------------------
// Kernel 6: out = pooled @ fc_w^T + fc_b   (64x10, trivial)
// ---------------------------------------------------------------------------
__global__ __launch_bounds__(256) void k_fc(const float* __restrict__ pooled,
                                            const float* __restrict__ fc_w,
                                            const float* __restrict__ fc_b,
                                            float* __restrict__ out) {
  const int t = blockIdx.x * 256 + threadIdx.x;
  if (t < Bb * Cc) {
    const int bb = t / Cc, c = t % Cc;
    float s = fc_b[c];
    for (int h = 0; h < Hh; ++h) s += pooled[(size_t)bb * Hh + h] * fc_w[(size_t)c * Hh + h];
    out[t] = s;
  }
}

// ---------------------------------------------------------------------------
extern "C" void kernel_launch(void* const* d_in, const int* in_sizes, int n_in,
                              void* d_out, int out_size, void* d_ws, size_t ws_size,
                              hipStream_t stream) {
  (void)in_sizes; (void)n_in; (void)out_size; (void)ws_size;
  const int*   x       = (const int*)  d_in[0];
  const float* embed_w = (const float*)d_in[1];
  const float* Wl      = (const float*)d_in[2];
  const float* bl      = (const float*)d_in[3];
  const float* Wsl     = (const float*)d_in[4];
  const float* bsl     = (const float*)d_in[5];
  const float* Wr      = (const float*)d_in[6];
  const float* br      = (const float*)d_in[7];
  const float* Wsr     = (const float*)d_in[8];
  const float* bsr     = (const float*)d_in[9];
  const float* conv_w  = (const float*)d_in[10];
  const float* conv_b  = (const float*)d_in[11];
  const float* fc_w    = (const float*)d_in[12];
  const float* fc_b    = (const float*)d_in[13];
  const float* cl0     = (const float*)d_in[14];
  const float* cr0     = (const float*)d_in[15];
  float* out = (float*)d_out;

  char* ws = (char*)d_ws;
  size_t off = 0;
  auto carve = [&](size_t bytes) -> char* {
    char* p = ws + off;
    off += (bytes + 255) & ~(size_t)255;
    return p;
  };
  const size_t BSD = (size_t)Bb * Ss * Dd;
  unsigned short* e_bf     = (unsigned short*)carve(BSD * 2);
  unsigned short* left_bf  = (unsigned short*)carve(BSD * 2);
  unsigned short* right_bf = (unsigned short*)carve(BSD * 2);
  float*          pre_l    = (float*)carve(BSD * 4);
  float*          pre_r    = (float*)carve(BSD * 4);
  unsigned short* Wl_bf    = (unsigned short*)carve((size_t)Dd * Dd * 2);
  unsigned short* Wsl_bf   = (unsigned short*)carve((size_t)Dd * Dd * 2);
  unsigned short* Wr_bf    = (unsigned short*)carve((size_t)Dd * Dd * 2);
  unsigned short* Wsr_bf   = (unsigned short*)carve((size_t)Dd * Dd * 2);
  unsigned short* convw_bf = (unsigned short*)carve((size_t)Hh * 3 * Dd * 2);
  float*          pooled   = (float*)carve((size_t)Bb * Hh * 4);

  k_prep<<<dim3((Hh * 3 * Dd + 255) / 256), 256, 0, stream>>>(
      Wl, Wsl, Wr, Wsr, conv_w, Wl_bf, Wsl_bf, Wr_bf, Wsr_bf, convw_bf, pooled);

  k_embed<<<dim3((unsigned)(BSD / 256)), 256, 0, stream>>>(x, embed_w, e_bf);

  k_pre<<<dim3(Bb * Ss / 64, 2), 256, 0, stream>>>(
      e_bf, Wsl_bf, Wsr_bf, bsl, bsr, bl, br, pre_l, pre_r);

  k_scan<<<dim3(2), 256, 0, stream>>>(
      Wl_bf, Wr_bf, pre_l, pre_r, cl0, cr0, left_bf, right_bf);

  k_conv<<<dim3(Bb * Ss / 64, 2), 256, 0, stream>>>(
      left_bf, e_bf, right_bf, convw_bf, conv_b, pooled);

  k_fc<<<dim3(3), 256, 0, stream>>>(pooled, fc_w, fc_b, out);
}
